// ConvOffset2D_7584912245430
// MI455X (gfx1250) — matrix operand order repaired and checked by reading, not compiled, not run on hardware
//
#include <hip/hip_runtime.h>
#include <math.h>

// ---------------------------------------------------------------------------
// Problem constants: B=8, H=W=128, C=256, 2C=512, K = 9*256 = 2304
// Conv as implicit GEMM: M = 131072 pixels, N = 512, K = 2304 (bf16 WMMA).
// ---------------------------------------------------------------------------

typedef __attribute__((ext_vector_type(16))) __bf16 v16bf;
typedef __attribute__((ext_vector_type(8)))  float  v8f;

union FragU {
    uint4 q[2];
    v16bf v;
};

// ---------------- fp32 -> bf16 conversion of x (33.5M elems) ----------------
__global__ __launch_bounds__(256) void cvt_x_kernel(const float* __restrict__ x,
                                                    __bf16* __restrict__ xb) {
    const int i = (blockIdx.x * 256 + threadIdx.x) * 4;
    float4 f = *(const float4*)&x[i];
    union { __bf16 h[4]; unsigned long long u; } u;
    u.h[0] = (__bf16)f.x; u.h[1] = (__bf16)f.y;
    u.h[2] = (__bf16)f.z; u.h[3] = (__bf16)f.w;
    *(unsigned long long*)&xb[i] = u.u;
}

// ------- kernel (3,3,256,512) fp32 -> wT[n][k] bf16, k = (kh*3+kw)*256+ci ---
__global__ __launch_bounds__(256) void cvt_w_kernel(const float* __restrict__ w,
                                                    __bf16* __restrict__ wT) {
    const int idx = blockIdx.x * 256 + threadIdx.x;  // 512*2304 total
    const int nn = idx / 2304;
    const int k  = idx - nn * 2304;
    wT[idx] = (__bf16)w[(size_t)k * 512 + nn];
}

// ---------------------------------------------------------------------------
// Implicit-GEMM conv: out[p, n] = sum_k A[p,k] * Wt[n,k] + bias[n]
// Tile: 128 pixels (one full image row) x 64 channels, BK = 32.
// 8 waves: (wm in 0..3) x (wn in 0..1); each wave owns a 32M x 32N sub-tile:
//   2 A-frags x 2 B-frags -> 4 v_wmma_f32_16x16x32_bf16 per K-step.
// Double-buffered LDS: one barrier per K-step; next-tile global loads issue
// before the WMMAs so their latency overlaps compute.
// ---------------------------------------------------------------------------
__global__ __launch_bounds__(256) void conv_offsets_kernel(
    const __bf16* __restrict__ xb, const __bf16* __restrict__ wT,
    const float* __restrict__ bias, float* __restrict__ off) {

    // row stride 40 bf16 (80B): conflict-friendly, keeps 16B alignment
    __shared__ __align__(16) __bf16 Al[2][128 * 40];  // 2 x 10240B
    __shared__ __align__(16) __bf16 Bl[2][64 * 40];   // 2 x  5120B

    const int tid = threadIdx.x;
    const int m0  = blockIdx.x * 128;       // pixel tile base (full row)
    const int n0  = blockIdx.y * 64;        // out-channel tile base
    const int b   = m0 >> 14;               // / (128*128)
    const int irow = (m0 & 16383) >> 7;     // image row; j runs 0..127

    // global->LDS loader mapping
    const int lrowA = tid >> 1;             // 0..127 pixel (== column j)
    const int lofsA = (tid & 1) << 4;       // 0 or 16 (bf16 elems)
    const int lrowB = tid >> 2;             // 0..63 out-channel
    const int lofsB = (tid & 3) << 3;       // 0,8,16,24

    // wave / fragment mapping (ISA 16-bit A 16x32, B 32x16, f32 C/D layouts)
    const int lane = tid & 31;
    const int wave = tid >> 5;
    const int wm   = wave >> 1;             // 0..3 -> m base = wm*32
    const int wn   = wave & 1;              // 0..1 -> n base = wn*32
    const int l16  = lane & 15;
    const int lhi  = lane >> 4;             // 0 or 1
    const int aoff0 = (wm * 32 + l16) * 40 + lhi * 8;
    const int aoff1 = aoff0 + 16 * 40;
    const int boff0 = (wn * 32 + l16) * 40 + lhi * 8;
    const int boff1 = boff0 + 16 * 40;

    v8f acc00 = {}, acc01 = {}, acc10 = {}, acc11 = {};

    uint4 aR0 = make_uint4(0, 0, 0, 0);
    uint4 aR1 = make_uint4(0, 0, 0, 0);
    uint4 bR  = make_uint4(0, 0, 0, 0);

    auto load_global = [&](int it) {
        const int t9 = it >> 3;             // 0..8  -> (kh,kw)
        const int kb = it & 7;              // ci block of 32
        const int ii = irow + (t9 / 3) - 1;
        const int jj = lrowA + (t9 % 3) - 1;
        if (((unsigned)ii < 128u) && ((unsigned)jj < 128u)) {
            const __bf16* p = &xb[((size_t)(((b << 7) + ii) << 7) + jj) * 256
                                  + (kb << 5) + lofsA];
            aR0 = *(const uint4*)p;
            aR1 = *(const uint4*)(p + 8);
        } else {
            aR0 = make_uint4(0, 0, 0, 0);
            aR1 = make_uint4(0, 0, 0, 0);
        }
        bR = *(const uint4*)&wT[(size_t)(n0 + lrowB) * 2304 + t9 * 256
                                + (kb << 5) + lofsB];
    };

    auto store_lds = [&](int buf) {
        *(uint4*)&Al[buf][lrowA * 40 + lofsA]     = aR0;
        *(uint4*)&Al[buf][lrowA * 40 + lofsA + 8] = aR1;
        *(uint4*)&Bl[buf][lrowB * 40 + lofsB]     = bR;
    };

    load_global(0);
    store_lds(0);
    __syncthreads();

    for (int it = 0; it < 72; ++it) {
        const int cur = it & 1;
        if (it + 1 < 72) load_global(it + 1);   // overlaps with WMMAs below

        FragU a0, a1, f0, f1;
        a0.q[0] = *(const uint4*)&Al[cur][aoff0];
        a0.q[1] = *(const uint4*)&Al[cur][aoff0 + 16];
        a1.q[0] = *(const uint4*)&Al[cur][aoff1];
        a1.q[1] = *(const uint4*)&Al[cur][aoff1 + 16];
        f0.q[0] = *(const uint4*)&Bl[cur][boff0];
        f0.q[1] = *(const uint4*)&Bl[cur][boff0 + 16];
        f1.q[0] = *(const uint4*)&Bl[cur][boff1];
        f1.q[1] = *(const uint4*)&Bl[cur][boff1 + 16];

        acc00 = __builtin_amdgcn_wmma_f32_16x16x32_bf16(
            false, a0.v, false, f0.v, (short)0, acc00, false, false);
        acc01 = __builtin_amdgcn_wmma_f32_16x16x32_bf16(
            false, a0.v, false, f1.v, (short)0, acc01, false, false);
        acc10 = __builtin_amdgcn_wmma_f32_16x16x32_bf16(
            false, a1.v, false, f0.v, (short)0, acc10, false, false);
        acc11 = __builtin_amdgcn_wmma_f32_16x16x32_bf16(
            false, a1.v, false, f1.v, (short)0, acc11, false, false);

        if (it + 1 < 72) store_lds((it + 1) & 1);  // other buffer: safe pre-barrier
        __syncthreads();
    }

    // Epilogue: D layout -> NHWC fp32 offsets (+bias)
    const int ng0 = n0 + wn * 32 + l16;
    const int ng1 = ng0 + 16;
    const float bs0 = bias[ng0];
    const float bs1 = bias[ng1];
    const int mb0 = m0 + wm * 32 + lhi * 8;        // A-frag 0 rows
    const int mb1 = mb0 + 16;                      // A-frag 1 rows
    #pragma unroll
    for (int v = 0; v < 8; ++v) {
        const size_t p0 = (size_t)(mb0 + v) * 512;
        const size_t p1 = (size_t)(mb1 + v) * 512;
        off[p0 + ng0] = acc00[v] + bs0;
        off[p0 + ng1] = acc01[v] + bs1;
        off[p1 + ng0] = acc10[v] + bs0;
        off[p1 + ng1] = acc11[v] + bs1;
    }
}

// ---------------------------------------------------------------------------
// Deformable bilinear sampling with the faithful keras reshape remap:
// offs[bc, n, k] <- offsets channel 2c+q at spatial r, where
//   t = 2n+k, q = t / (H*W), r = t % (H*W).
// x (134MB) fits in the 192MB L2, so the 4 gathers are mostly L2-resident.
// ---------------------------------------------------------------------------
__global__ __launch_bounds__(256) void deform_sample_kernel(
    const float* __restrict__ x, const float* __restrict__ off,
    float* __restrict__ out) {
    const int idx = blockIdx.x * 256 + threadIdx.x;   // 33554432 total
    const int c  = idx & 255;
    const int n  = (idx >> 8) & 16383;
    const int b  = idx >> 22;
    const int i2 = n >> 7;
    const int j2 = n & 127;

    const int t0 = n << 1;
    const int t1 = t0 | 1;
    const int q0 = t0 >> 14, r0 = t0 & 16383;
    const int q1 = t1 >> 14, r1 = t1 & 16383;
    const size_t obase = (size_t)b * 16384;
    const float o0 = off[(obase + r0) * 512 + (c << 1) + q0];
    const float o1 = off[(obase + r1) * 512 + (c << 1) + q1];

    const float c0 = fminf(fmaxf(o0 + (float)i2, 0.0f), 127.0f);
    const float c1 = fminf(fmaxf(o1 + (float)j2, 0.0f), 127.0f);
    const int lt0 = (int)floorf(c0), rb0 = (int)ceilf(c0);
    const int lt1 = (int)floorf(c1), rb1 = (int)ceilf(c1);
    const float f0 = c0 - (float)lt0;
    const float f1 = c1 - (float)lt1;

    const size_t xbase = ((size_t)b * 16384) * 256 + c;
    const float vlt = x[xbase + (size_t)(lt0 * 128 + lt1) * 256];
    const float vrt = x[xbase + (size_t)(rb0 * 128 + lt1) * 256];
    const float vlb = x[xbase + (size_t)(lt0 * 128 + rb1) * 256];
    const float vrb = x[xbase + (size_t)(rb0 * 128 + rb1) * 256];
    const float vt = vlt + (vrt - vlt) * f0;
    const float vb = vlb + (vrb - vlb) * f0;
    out[idx] = vt + (vb - vt) * f1;
}

// ---------------------------------------------------------------------------
extern "C" void kernel_launch(void* const* d_in, const int* in_sizes, int n_in,
                              void* d_out, int out_size, void* d_ws, size_t ws_size,
                              hipStream_t stream) {
    const float* x    = (const float*)d_in[0];
    const float* kern = (const float*)d_in[1];
    const float* bias = (const float*)d_in[2];
    float* out = (float*)d_out;

    // Workspace layout:
    //   [0, 268435456)                : offsets fp32  (B*H*W*512*4)
    //   [268435456, +67108864)        : x bf16        (B*H*W*256*2)
    //   [335544320, +2359296)         : wT bf16       (512*2304*2)
    char* ws = (char*)d_ws;
    float*  off = (float*)ws;
    __bf16* xb  = (__bf16*)(ws + (size_t)268435456);
    __bf16* wT  = (__bf16*)(ws + (size_t)268435456 + (size_t)67108864);

    cvt_x_kernel<<<32768, 256, 0, stream>>>(x, xb);           // 33554432 / (256*4)
    cvt_w_kernel<<<4608, 256, 0, stream>>>(kern, wT);         // 1179648 / 256
    conv_offsets_kernel<<<dim3(1024, 8), 256, 0, stream>>>(xb, wT, bias, off);
    deform_sample_kernel<<<131072, 256, 0, stream>>>(x, off, out);
}
